// CustomAttentionWithPE_90202903150965
// MI455X (gfx1250) — matrix operand order repaired and checked by reading, not compiled, not run on hardware
//
#include <hip/hip_runtime.h>

typedef __attribute__((ext_vector_type(16))) __bf16 v16bf;
typedef __attribute__((ext_vector_type(8)))  float  v8f;
typedef __attribute__((ext_vector_type(4)))  int    i32x4;

#define SEQ 2048
#define DM  1024
#define NH  16
#define HD  64

union AFrag {
  v16bf v;
  uint4 q[2];
};

__device__ __forceinline__ unsigned short f2bf(float f) {
  union { float f; unsigned u; } x; x.f = f;
  return (unsigned short)((x.u + 0x7FFFu + ((x.u >> 16) & 1u)) >> 16);
}

__device__ __forceinline__ v8f bf16_wmma(const AFrag& a, const AFrag& b, v8f c) {
  return __builtin_amdgcn_wmma_f32_16x16x32_bf16(false, a.v, false, b.v,
                                                 (short)0, c, false, false);
}

// ---------------------------------------------------------------------------
// CDNA5 async global->LDS copy (ASYNCcnt-tracked). Falls back to plain copy.
// Builtin signature (from toolchain diagnostic): (int4 AS1*, int4 AS3*, Ii, Ii)
// ---------------------------------------------------------------------------
#if defined(__gfx1250__) && __has_builtin(__builtin_amdgcn_global_load_async_to_lds_b128)
#define USE_ASYNC_LDS 1
#else
#define USE_ASYNC_LDS 0
#endif

__device__ __forceinline__ void cp16(unsigned short* l, const unsigned short* g) {
#if USE_ASYNC_LDS
  typedef __attribute__((address_space(1))) i32x4* gp128;
  typedef __attribute__((address_space(3))) i32x4* lp128;
  __builtin_amdgcn_global_load_async_to_lds_b128(
      (gp128)(unsigned long long)(uintptr_t)g,
      (lp128)(unsigned)(uintptr_t)l, 0, 0);
#else
  *(uint4*)l = *(const uint4*)g;
#endif
}

__device__ __forceinline__ void wait_async() {
#if USE_ASYNC_LDS
#if __has_builtin(__builtin_amdgcn_s_wait_asynccnt)
  __builtin_amdgcn_s_wait_asynccnt(0);
#else
  asm volatile("s_wait_asynccnt 0x0" ::: "memory");
#endif
#endif
}

// ---------------------------------------------------------------------------
// Kernel 0a: flat f32 -> bf16 convert (for hidden_states).
// ---------------------------------------------------------------------------
__global__ __launch_bounds__(256) void convert_bf16_kernel(
    const float* __restrict__ src, unsigned short* __restrict__ dst)
{
  int i = (blockIdx.x * 256 + threadIdx.x) * 4;
  float4 f = *(const float4*)(src + i);
  *(ushort4*)(dst + i) = make_ushort4(f2bf(f.x), f2bf(f.y), f2bf(f.z), f2bf(f.w));
}

// ---------------------------------------------------------------------------
// Kernel 0b: transpose + convert: src [K][N] f32 -> dst [N][K] bf16.
// ---------------------------------------------------------------------------
__global__ __launch_bounds__(256) void transpose_convert_kernel(
    const float* __restrict__ src, unsigned short* __restrict__ dst,
    int K, int N)
{
  __shared__ float tile[32][33];
  const int k0 = blockIdx.y * 32, n0 = blockIdx.x * 32;
  const int r = threadIdx.x >> 3, c = (threadIdx.x & 7) * 4;
  float4 f = *(const float4*)(src + (size_t)(k0 + r) * N + n0 + c);
  tile[r][c + 0] = f.x; tile[r][c + 1] = f.y;
  tile[r][c + 2] = f.z; tile[r][c + 3] = f.w;
  __syncthreads();
  *(ushort4*)(dst + (size_t)(n0 + r) * K + k0 + c) =
      make_ushort4(f2bf(tile[c + 0][r]), f2bf(tile[c + 1][r]),
                   f2bf(tile[c + 2][r]), f2bf(tile[c + 3][r]));
}

// ---------------------------------------------------------------------------
// Shared GEMM machinery: A [M][1024] bf16, B^T [N][1024] bf16.
// Block tile 64(M) x 256(N), 8 waves (2x4), K-chunks of 32, double buffered.
// ---------------------------------------------------------------------------
__device__ __forceinline__ void gemm_stage(
    const unsigned short* __restrict__ A, const unsigned short* __restrict__ BT,
    int Mbase, int Nbase, int kb, int tid,
    unsigned short* lA, unsigned short* lB)
{
  const int ar = tid >> 2, ac = (tid & 3) * 8;
  cp16(&lA[ar * 32 + ac], A + (size_t)(Mbase + ar) * DM + kb + ac);
  const unsigned short* g = BT + (size_t)(Nbase + tid) * DM + kb;
  unsigned short* l = &lB[tid * 32];
  cp16(l, g);
  cp16(l + 8,  g + 8);
  cp16(l + 16, g + 16);
  cp16(l + 24, g + 24);
}

__device__ __forceinline__ void gemm_compute(
    const unsigned short* lA, const unsigned short* lB,
    int wm, int wn, int l15, int hi, v8f acc[2][4])
{
  AFrag a[2], b[4];
#pragma unroll
  for (int mt = 0; mt < 2; ++mt) {
    const unsigned short* p = &lA[(wm * 32 + mt * 16 + l15) * 32 + hi * 8];
    a[mt].q[0] = *(const uint4*)p;
    a[mt].q[1] = *(const uint4*)(p + 16);
  }
#pragma unroll
  for (int nt = 0; nt < 4; ++nt) {
    const unsigned short* p = &lB[(wn * 64 + nt * 16 + l15) * 32 + hi * 8];
    b[nt].q[0] = *(const uint4*)p;
    b[nt].q[1] = *(const uint4*)(p + 16);
  }
#pragma unroll
  for (int nt = 0; nt < 4; ++nt)
#pragma unroll
    for (int mt = 0; mt < 2; ++mt)
      acc[mt][nt] = bf16_wmma(a[mt], b[nt], acc[mt][nt]);
}

// ---------------------------------------------------------------------------
// Kernel 1: QKV GEMM (bf16 in, fp32 acc) + bias + RoPE, scatter Q/K/V^T bf16.
// ---------------------------------------------------------------------------
__global__ __launch_bounds__(256) void qkv_rope_kernel(
    const unsigned short* __restrict__ Xb, const unsigned short* __restrict__ WT,
    const float* __restrict__ bias,
    unsigned short* __restrict__ Qb, unsigned short* __restrict__ Kb,
    unsigned short* __restrict__ Vtb)
{
  __shared__ unsigned short lA[2][64 * 32];
  __shared__ unsigned short lB[2][256 * 32];
  const int tid  = threadIdx.x;
  const int lane = tid & 31, w = tid >> 5;
  const int l15  = lane & 15, hi = lane >> 4;
  const int wm   = w >> 2, wn = w & 3;
  const int Mbase = blockIdx.y * 64;
  const int Nbase = blockIdx.x * 256;
  v8f acc[2][4] = {};

  gemm_stage(Xb, WT, Mbase, Nbase, 0, tid, lA[0], lB[0]);
  for (int kb = 0; kb < DM; kb += 32) {
    const int cur = (kb >> 5) & 1;
    wait_async();
    __syncthreads();
    if (kb + 32 < DM)
      gemm_stage(Xb, WT, Mbase, Nbase, kb + 32, tid, lA[cur ^ 1], lB[cur ^ 1]);
    gemm_compute(lA[cur], lB[cur], wm, wn, l15, hi, acc);
  }

  // Epilogue: bias + RoPE + scatter. Wave owns a 64-wide head-aligned N span,
  // so sel (q/k/v) and head h are wave-uniform; RoPE partner d+32 is two
  // N-tiles over in the SAME lane and VGPR index.
  const int nwave = Nbase + wn * 64;
  const int sel = nwave / DM;
  const int h   = (nwave % DM) / HD;
  float bv[4];
#pragma unroll
  for (int nt = 0; nt < 4; ++nt) bv[nt] = bias[nwave + nt * 16 + l15];

#pragma unroll
  for (int mt = 0; mt < 2; ++mt) {
#pragma unroll
    for (int r = 0; r < 8; ++r) {
      int m    = Mbase + wm * 32 + mt * 16 + r + 8 * hi;
      int bidx = m >> 11;          // / SEQ
      int spos = m & (SEQ - 1);
      if (sel == 2) {
#pragma unroll
        for (int nt = 0; nt < 4; ++nt) {
          int d = nt * 16 + l15;
          float v = acc[mt][nt][r] + bv[nt];
          Vtb[((size_t)(bidx * NH + h) * HD + d) * SEQ + spos] = f2bf(v);
        }
      } else {
        unsigned short* dst = (sel == 0) ? Qb : Kb;
        size_t base = ((size_t)(bidx * NH + h) * SEQ + spos) * HD;
#pragma unroll
        for (int nt = 0; nt < 2; ++nt) {
          int d = nt * 16 + l15;                         // 0..31
          float invf = __powf(10000.0f, -(float)d * (1.0f / 32.0f));
          float ang  = (float)spos * invf;
          float sn, cs; __sincosf(ang, &sn, &cs);
          float x1 = acc[mt][nt][r]     + bv[nt];
          float x2 = acc[mt][nt + 2][r] + bv[nt + 2];
          dst[base + d]      = f2bf(x1 * cs - x2 * sn);
          dst[base + d + 32] = f2bf(x2 * cs + x1 * sn);
        }
      }
    }
  }
}

// ---------------------------------------------------------------------------
// Kernel 2: flash attention. One WG (4 waves) per (b,h, 64 query rows);
// each wave owns 16 query rows, loops over key tiles of 32 with online softmax.
// ---------------------------------------------------------------------------
__global__ __launch_bounds__(128) void flash_attn_kernel(
    const unsigned short* __restrict__ Qb, const unsigned short* __restrict__ Kb,
    const unsigned short* __restrict__ Vtb, unsigned short* __restrict__ Ob)
{
  __shared__ unsigned short lP[4][16 * 32];   // per-wave P staging (C->A relayout)
  const int tid = threadIdx.x, lane = tid & 31, w = tid >> 5;
  const int l15 = lane & 15, hi = lane >> 4;
  const int bh = blockIdx.y;
  const int b  = bh >> 4;
  const int h  = bh & 15;
  const int q0 = blockIdx.x * 64 + w * 16;
  const unsigned short* Qp = Qb  + (size_t)bh * SEQ * HD;
  const unsigned short* Kp = Kb  + (size_t)bh * SEQ * HD;
  const unsigned short* Vp = Vtb + (size_t)bh * HD * SEQ;

  AFrag qf[2];
  {
    const unsigned short* p = Qp + (size_t)(q0 + l15) * HD;
#pragma unroll
    for (int c = 0; c < 2; ++c) {
      int k0 = c * 32 + hi * 8;
      qf[c].q[0] = *(const uint4*)(p + k0);
      qf[c].q[1] = *(const uint4*)(p + k0 + 16);
    }
  }
  v8f o[4] = {};
  float rm[8], rs[8];
#pragma unroll
  for (int r = 0; r < 8; ++r) { rm[r] = -3.0e38f; rs[r] = 0.0f; }

  for (int kb = 0; kb < SEQ; kb += 32) {
    v8f s[2] = {};
#pragma unroll
    for (int nt = 0; nt < 2; ++nt) {
      const unsigned short* p = Kp + (size_t)(kb + nt * 16 + l15) * HD;
#pragma unroll
      for (int c = 0; c < 2; ++c) {
        AFrag bf;
        int k0 = c * 32 + hi * 8;
        bf.q[0] = *(const uint4*)(p + k0);
        bf.q[1] = *(const uint4*)(p + k0 + 16);
        s[nt] = bf16_wmma(qf[c], bf, s[nt]);
      }
    }
    // online softmax over the 32-key tile (rows live as VGPR r + lane group)
    float alpha[8];
#pragma unroll
    for (int r = 0; r < 8; ++r) {
      float v0 = s[0][r] * 0.125f;     // 1/sqrt(64)
      float v1 = s[1][r] * 0.125f;
      float mx = fmaxf(v0, v1);
#pragma unroll
      for (int off = 8; off > 0; off >>= 1)
        mx = fmaxf(mx, __shfl_xor(mx, off, 32));
      float mnew = fmaxf(rm[r], mx);
      float a  = __expf(rm[r] - mnew);
      float p0 = __expf(v0 - mnew);
      float p1 = __expf(v1 - mnew);
      float sum = p0 + p1;
#pragma unroll
      for (int off = 8; off > 0; off >>= 1)
        sum += __shfl_xor(sum, off, 32);
      rm[r] = mnew;
      rs[r] = rs[r] * a + sum;
      alpha[r] = a;
      s[0][r] = p0;
      s[1][r] = p1;
    }
#pragma unroll
    for (int vt = 0; vt < 4; ++vt)
#pragma unroll
      for (int r = 0; r < 8; ++r)
        o[vt][r] *= alpha[r];

    // C-layout -> A-layout relayout of P through LDS
    __syncthreads();
    {
      unsigned short* lp = lP[w];
#pragma unroll
      for (int r = 0; r < 8; ++r) {
        int m = r + 8 * hi;
        lp[m * 32 + l15]      = f2bf(s[0][r]);
        lp[m * 32 + 16 + l15] = f2bf(s[1][r]);
      }
    }
    __syncthreads();
    AFrag pf;
    {
      const unsigned short* p = &lP[w][l15 * 32 + hi * 8];
      pf.q[0] = *(const uint4*)p;
      pf.q[1] = *(const uint4*)(p + 16);
    }
#pragma unroll
    for (int vt = 0; vt < 4; ++vt) {
      AFrag vf;
      const unsigned short* p = Vp + (size_t)(vt * 16 + l15) * SEQ + kb + hi * 8;
      vf.q[0] = *(const uint4*)p;
      vf.q[1] = *(const uint4*)(p + 16);
      o[vt] = bf16_wmma(pf, vf, o[vt]);
    }
  }

  // finalize: divide by row sums, store [B,S,H*64] bf16 for the output GEMM
#pragma unroll
  for (int r = 0; r < 8; ++r) {
    float inv = 1.0f / rs[r];
    int srow = q0 + r + 8 * hi;
    size_t base = ((size_t)(b * SEQ + srow)) * DM + h * HD;
#pragma unroll
    for (int vt = 0; vt < 4; ++vt)
      Ob[base + vt * 16 + l15] = f2bf(o[vt][r] * inv);
  }
}

// ---------------------------------------------------------------------------
// Kernel 3: output projection (bf16 A, bf16 W_o^T) + bias, fp32 out.
// ---------------------------------------------------------------------------
__global__ __launch_bounds__(256) void out_proj_kernel(
    const unsigned short* __restrict__ Ab, const unsigned short* __restrict__ WT,
    const float* __restrict__ bias, float* __restrict__ out)
{
  __shared__ unsigned short lA[2][64 * 32];
  __shared__ unsigned short lB[2][256 * 32];
  const int tid  = threadIdx.x;
  const int lane = tid & 31, w = tid >> 5;
  const int l15  = lane & 15, hi = lane >> 4;
  const int wm   = w >> 2, wn = w & 3;
  const int Mbase = blockIdx.y * 64;
  const int Nbase = blockIdx.x * 256;
  v8f acc[2][4] = {};

  gemm_stage(Ab, WT, Mbase, Nbase, 0, tid, lA[0], lB[0]);
  for (int kb = 0; kb < DM; kb += 32) {
    const int cur = (kb >> 5) & 1;
    wait_async();
    __syncthreads();
    if (kb + 32 < DM)
      gemm_stage(Ab, WT, Mbase, Nbase, kb + 32, tid, lA[cur ^ 1], lB[cur ^ 1]);
    gemm_compute(lA[cur], lB[cur], wm, wn, l15, hi, acc);
  }

  float bv[4];
#pragma unroll
  for (int nt = 0; nt < 4; ++nt) bv[nt] = bias[Nbase + wn * 64 + nt * 16 + l15];
#pragma unroll
  for (int mt = 0; mt < 2; ++mt)
#pragma unroll
    for (int r = 0; r < 8; ++r) {
      int m = Mbase + wm * 32 + mt * 16 + r + 8 * hi;
#pragma unroll
      for (int nt = 0; nt < 4; ++nt)
        out[(size_t)m * DM + Nbase + wn * 64 + nt * 16 + l15] =
            acc[mt][nt][r] + bv[nt];
    }
}

// ---------------------------------------------------------------------------
extern "C" void kernel_launch(void* const* d_in, const int* in_sizes, int n_in,
                              void* d_out, int out_size, void* d_ws, size_t ws_size,
                              hipStream_t stream) {
  (void)in_sizes; (void)n_in; (void)out_size; (void)ws_size;
  const float* X    = (const float*)d_in[0];   // [2,2048,1024]
  const float* Wqkv = (const float*)d_in[1];   // [1024,3072]
  const float* bqkv = (const float*)d_in[2];   // [3072]
  const float* Wo   = (const float*)d_in[3];   // [1024,1024]
  const float* bo   = (const float*)d_in[4];   // [1024]
  float* out = (float*)d_out;                  // [2,2048,1024] f32

  const size_t qkvE = (size_t)2 * NH * SEQ * HD;   // 4 Mi elems
  unsigned short* Qb   = (unsigned short*)d_ws;
  unsigned short* Kb   = Qb + qkvE;
  unsigned short* Vtb  = Kb + qkvE;
  unsigned short* Ob   = Vtb + qkvE;
  unsigned short* Xbf  = Ob + qkvE;                         // 4 Mi
  unsigned short* WqT  = Xbf + (size_t)2 * SEQ * DM;        // 3 Mi
  unsigned short* WoT  = WqT + (size_t)3 * DM * DM;         // 1 Mi

  // precision/layout prep (off the critical path; ~50 MB total traffic)
  convert_bf16_kernel<<<dim3(4096), 256, 0, stream>>>(X, Xbf);
  transpose_convert_kernel<<<dim3(96, 32), 256, 0, stream>>>(Wqkv, WqT, DM, 3 * DM);
  transpose_convert_kernel<<<dim3(32, 32), 256, 0, stream>>>(Wo, WoT, DM, DM);

  // QKV GEMM: M=4096 (64/blk), N=3072 (256/blk)
  qkv_rope_kernel<<<dim3(12, 64), 256, 0, stream>>>(Xbf, WqT, bqkv, Qb, Kb, Vtb);
  // attention: 32 q-blocks x 32 (b,h)
  flash_attn_kernel<<<dim3(32, 32), 128, 0, stream>>>(Qb, Kb, Vtb, Ob);
  // output GEMM: M=4096, N=1024 (256/blk)
  out_proj_kernel<<<dim3(4, 64), 256, 0, stream>>>(Ob, WoT, bo, out);
}
